// MultiHeadAttentionLayer_27659589386281
// MI455X (gfx1250) — compile-verified, matrix operand order checked by reading, not run on hardware
//
#include <hip/hip_runtime.h>

#define MODEL_DIM 1024
#define N_HEADS   16
#define HEAD_DIM  64
#define BATCH     4
#define SEQ       2048
#define BS        (BATCH * SEQ)   // 8192

typedef __attribute__((ext_vector_type(16))) _Float16 v16h;
typedef __attribute__((ext_vector_type(8)))  float    v8f;
typedef __attribute__((ext_vector_type(4)))  int      vi4;

// ---------------------------------------------------------------------------
// Async global->LDS copy (gfx1250 GLOBAL_LOAD_ASYNC_TO_LDS_B128, ASYNCcnt).
// Builtin expects int4* in global/LDS address spaces.
// ---------------------------------------------------------------------------
#if defined(__has_builtin)
#if __has_builtin(__builtin_amdgcn_global_load_async_to_lds_b128)
#define HAVE_ASYNC_LDS 1
#endif
#endif

typedef __attribute__((address_space(1))) vi4* as1_vi4p;
typedef __attribute__((address_space(3))) vi4* as3_vi4p;

__device__ __forceinline__ void async_cp16(_Float16* lds_dst, const _Float16* gsrc) {
#if defined(HAVE_ASYNC_LDS)
  __builtin_amdgcn_global_load_async_to_lds_b128(
      (as1_vi4p)(void*)gsrc, (as3_vi4p)(void*)lds_dst,
      /*imm offset=*/0, /*cpol=*/0);
#else
  *reinterpret_cast<uint4*>(lds_dst) = *reinterpret_cast<const uint4*>(gsrc);
#endif
}

template <int N>
__device__ __forceinline__ void wait_asynccnt() {
#if defined(__has_builtin)
#if __has_builtin(__builtin_amdgcn_s_wait_asynccnt)
  __builtin_amdgcn_s_wait_asynccnt(N);
#else
  asm volatile("s_wait_asynccnt %0" ::"i"(N));
#endif
#else
  asm volatile("s_wait_asynccnt %0" ::"i"(N));
#endif
}

// ---------------------------------------------------------------------------
// WMMA wrapper: D = A(16x32 f16) * B(32x16 f16) + C(16x16 f32)
// ---------------------------------------------------------------------------
__device__ __forceinline__ v8f wmma_f16(v16h a, v16h b, v8f c) {
  return __builtin_amdgcn_wmma_f32_16x16x32_f16(
      /*neg_a=*/false, a, /*neg_b=*/false, b,
      /*c_mod=*/(short)0, c, /*reuse_a=*/false, /*reuse_b=*/false);
}

// ---------------------------------------------------------------------------
// Load a 16x32 f16 A-fragment (also valid for B^T = [N=16,K=32] row-major).
// ISA 7.12.2: lane r=lane&15 holds row r; halves 0..7 = K kb..kb+7,
// halves 8..15 = K 16+kb..16+kb+7, kb = (lane&16) ? 8 : 0.
// ---------------------------------------------------------------------------
__device__ __forceinline__ v16h frag_ld(const _Float16* p0, int stride, int lane) {
  const int r  = lane & 15;
  const int kb = (lane & 16) ? 8 : 0;
  const _Float16* p = p0 + r * stride + kb;
  v16h f;
#pragma unroll
  for (int i = 0; i < 8; ++i) f[i] = p[i];
#pragma unroll
  for (int i = 0; i < 8; ++i) f[8 + i] = p[16 + i];
  return f;
}

// Synchronous 8-element chunk -> LDS with f32->f16 conversion (activations).
__device__ __forceinline__ void stage8(_Float16* dst, const float* s) {
  const float4 a = *reinterpret_cast<const float4*>(s);
  const float4 b = *reinterpret_cast<const float4*>(s + 4);
  alignas(16) _Float16 t[8];
  t[0] = (_Float16)a.x; t[1] = (_Float16)a.y; t[2] = (_Float16)a.z; t[3] = (_Float16)a.w;
  t[4] = (_Float16)b.x; t[5] = (_Float16)b.y; t[6] = (_Float16)b.z; t[7] = (_Float16)b.w;
  *reinterpret_cast<uint4*>(dst) = *reinterpret_cast<const uint4*>(t);
}
// f16 source: async DMA, no VGPR round-trip.
__device__ __forceinline__ void stage8(_Float16* dst, const _Float16* s) {
  async_cp16(dst, s);
}

// ---------------------------------------------------------------------------
// GEMM: C[M,N] = A[M,K] * Bw[N,K]^T   (A in TA, Bw f16, lda = ldb = K)
// Block tile 128x128, K-step 32, 256 threads = 8 waves (2x4), each wave 64x32.
// Double-buffered LDS; B (and f16 A) staged with async global->LDS copies.
// TRANS_V: scatter-store f16 result as V^T [B,H,HEAD_DIM,SEQ].
// OUT_F16: store f16 row-major, else f32 row-major.
// ---------------------------------------------------------------------------
template <typename TA, bool TRANS_V, bool OUT_F16>
__global__ void __launch_bounds__(256)
gemm_wmma(const TA* __restrict__ A, const _Float16* __restrict__ Bw,
          void* __restrict__ C, int K, int ldc) {
  __shared__ _Float16 As[2][128 * 32];
  __shared__ _Float16 Bs[2][128 * 32];

  const int tid  = threadIdx.x;
  const int lane = tid & 31;
  const int wave = tid >> 5;
  const int wm   = wave >> 2;   // 0..1  -> 64-row slab
  const int wn   = wave & 3;    // 0..3  -> 32-col slab
  const int m0   = blockIdx.y * 128;
  const int n0   = blockIdx.x * 128;

  // async instructions per wave per stage: 2 for B, +2 for A when A is f16.
  constexpr int NASYNC = (sizeof(TA) == 2) ? 4 : 2;

  v8f zero = {};
  v8f acc[4][2];
#pragma unroll
  for (int i = 0; i < 4; ++i)
#pragma unroll
    for (int j = 0; j < 2; ++j) acc[i][j] = zero;

  auto stage = [&](int buf, int k0) {
    for (int c = tid; c < 512; c += 256) {
      const int row = c >> 2;
      const int off = (c & 3) * 8;
      stage8(&As[buf][row * 32 + off], A + (long)(m0 + row) * K + k0 + off);
      async_cp16(&Bs[buf][row * 32 + off], Bw + (long)(n0 + row) * K + k0 + off);
    }
  };

  const int KT = K >> 5;
  stage(0, 0);
  for (int kt = 0; kt < KT; ++kt) {
    const int cur = kt & 1;
    if (kt + 1 < KT) {
      stage(cur ^ 1, (kt + 1) << 5);
      if (kt + 2 < KT)  // gfx1250 global_prefetch_b8 of the tile after next
        __builtin_prefetch(A + (long)(m0 + (tid >> 1)) * K + ((kt + 2) << 5), 0, 1);
      wait_asynccnt<NASYNC>();   // tile `cur` async copies complete
    } else {
      wait_asynccnt<0>();
    }
    __syncthreads();

    v16h fa[4], fb[2];
#pragma unroll
    for (int i = 0; i < 4; ++i)
      fa[i] = frag_ld(&As[cur][(wm * 64 + 16 * i) * 32], 32, lane);
#pragma unroll
    for (int j = 0; j < 2; ++j)
      fb[j] = frag_ld(&Bs[cur][(wn * 32 + 16 * j) * 32], 32, lane);

#pragma unroll
    for (int i = 0; i < 4; ++i)
#pragma unroll
      for (int j = 0; j < 2; ++j)
        acc[i][j] = wmma_f16(fa[i], fb[j], acc[i][j]);

    __syncthreads();
  }

  // Epilogue. C-fragment element (m,n): m = r + 8*(lane>>4), n = lane&15.
  const int mhi = (lane >> 4) * 8;
  const int nl  = lane & 15;
#pragma unroll
  for (int i = 0; i < 4; ++i) {
#pragma unroll
    for (int j = 0; j < 2; ++j) {
#pragma unroll
      for (int r = 0; r < 8; ++r) {
        const int   mg = m0 + wm * 64 + 16 * i + r + mhi;
        const int   ng = n0 + wn * 32 + 16 * j + nl;
        const float vv = acc[i][j][r];
        if (TRANS_V) {
          const int hh = ng >> 6, dd = ng & (HEAD_DIM - 1);
          const int bb = mg >> 11, ss = mg & (SEQ - 1);
          ((_Float16*)C)[(((long)bb * N_HEADS + hh) * HEAD_DIM + dd) * SEQ + ss] =
              (_Float16)vv;
        } else if (OUT_F16) {
          ((_Float16*)C)[(long)mg * ldc + ng] = (_Float16)vv;
        } else {
          ((float*)C)[(long)mg * ldc + ng] = vv;
        }
      }
    }
  }
}

// ---------------------------------------------------------------------------
// Weight conversion: f32 -> f16 (one-shot, 1M elements per weight matrix).
// ---------------------------------------------------------------------------
__global__ void __launch_bounds__(256)
cvt_f32_f16(const float* __restrict__ src, _Float16* __restrict__ dst, int n) {
  const int i = (blockIdx.x * 256 + threadIdx.x) * 8;
  if (i < n) {
    const float4 a = *reinterpret_cast<const float4*>(&src[i]);
    const float4 b = *reinterpret_cast<const float4*>(&src[i + 4]);
    alignas(16) _Float16 t[8];
    t[0] = (_Float16)a.x; t[1] = (_Float16)a.y; t[2] = (_Float16)a.z; t[3] = (_Float16)a.w;
    t[4] = (_Float16)b.x; t[5] = (_Float16)b.y; t[6] = (_Float16)b.z; t[7] = (_Float16)b.w;
    *reinterpret_cast<uint4*>(&dst[i]) = *reinterpret_cast<const uint4*>(t);
  }
}

// ---------------------------------------------------------------------------
// Flash-style attention. Block = 256 threads = 8 waves, one (b,h) per
// blockIdx.y; each wave owns a 16-row Q tile (block covers 128 Q rows).
// K/V tiles double-buffered via async global->LDS DMA.
// qh/kh: [B,S,H*dh] f16.  vT: [B,H,dh,S] f16.  ctx out: [B,S,H*dh] f16.
// ---------------------------------------------------------------------------
__global__ void __launch_bounds__(256)
attn_kernel(const _Float16* __restrict__ qh, const _Float16* __restrict__ kh,
            const _Float16* __restrict__ vT, _Float16* __restrict__ ctx) {
  __shared__ _Float16 kbuf[2][32 * 64];    // K tile   [sk=32][d=64]
  __shared__ _Float16 vbuf[2][64 * 32];    // V^T tile [d=64][sk=32]
  __shared__ _Float16 pbuf[8][16 * 32];    // per-wave P staging

  const int tid  = threadIdx.x;
  const int lane = tid & 31;
  const int wave = tid >> 5;
  const int bh   = blockIdx.y;             // b*16 + h
  const int b    = bh >> 4;
  const int h    = bh & 15;
  const int q0   = blockIdx.x * 128 + wave * 16;

  // Q fragments: 16 rows x 64 K, as two 16x32 A-fragments (stride = MODEL_DIM).
  const _Float16* qbase = qh + ((long)(b * SEQ + q0)) * MODEL_DIM + h * HEAD_DIM;
  const v16h aq0 = frag_ld(qbase, MODEL_DIM, lane);
  const v16h aq1 = frag_ld(qbase + 32, MODEL_DIM, lane);

  v8f zero = {};
  v8f acc[4];
  float rm[8], rl[8];
#pragma unroll
  for (int j = 0; j < 4; ++j) acc[j] = zero;
#pragma unroll
  for (int r = 0; r < 8; ++r) { rm[r] = -1e30f; rl[r] = 0.0f; }

  const _Float16* kbase_g = kh + (long)b * SEQ * MODEL_DIM + h * HEAD_DIM;
  const _Float16* vbase_g = vT + (long)bh * HEAD_DIM * SEQ;

  const int krow = tid >> 3, koff = (tid & 7) * 8;   // 256 chunks of K tile
  const int vrow = tid >> 2, voff = (tid & 3) * 8;   // 256 chunks of V tile

  auto stage = [&](int buf, int sk0) {
    async_cp16(&kbuf[buf][krow * 64 + koff],
               kbase_g + (long)(sk0 + krow) * MODEL_DIM + koff);
    async_cp16(&vbuf[buf][vrow * 32 + voff],
               vbase_g + (long)vrow * SEQ + sk0 + voff);
  };

  const int SKT = SEQ / 32;   // 64 steps
  stage(0, 0);
  for (int skt = 0; skt < SKT; ++skt) {
    const int cur = skt & 1;
    if (skt + 1 < SKT) {
      stage(cur ^ 1, (skt + 1) * 32);
      wait_asynccnt<2>();      // current tile's 2 async copies are complete
    } else {
      wait_asynccnt<0>();
    }
    __syncthreads();

    // Scores: two 16x16 tiles (sk columns 0..15 and 16..31), K-dim = dh = 64.
    v8f s0 = zero, s1 = zero;
    s0 = wmma_f16(aq0, frag_ld(&kbuf[cur][0], 64, lane), s0);
    s0 = wmma_f16(aq1, frag_ld(&kbuf[cur][32], 64, lane), s0);
    s1 = wmma_f16(aq0, frag_ld(&kbuf[cur][16 * 64], 64, lane), s1);
    s1 = wmma_f16(aq1, frag_ld(&kbuf[cur][16 * 64 + 32], 64, lane), s1);

    // Online softmax. Row m = r + 8*(lane>>4) lives across a 16-lane half.
    float t[8], sc[8], p0[8], p1[8], ps[8];
#pragma unroll
    for (int r = 0; r < 8; ++r) {
      s0[r] *= 0.125f;                       // 1/sqrt(HEAD_DIM)
      s1[r] *= 0.125f;
      t[r] = fmaxf(s0[r], s1[r]);
    }
#pragma unroll
    for (int m = 1; m < 16; m <<= 1)
#pragma unroll
      for (int r = 0; r < 8; ++r) t[r] = fmaxf(t[r], __shfl_xor(t[r], m, 32));
#pragma unroll
    for (int r = 0; r < 8; ++r) {
      const float mn = fmaxf(rm[r], t[r]);
      sc[r] = __expf(rm[r] - mn);
      rm[r] = mn;
      p0[r] = __expf(s0[r] - mn);
      p1[r] = __expf(s1[r] - mn);
      ps[r] = p0[r] + p1[r];
    }
#pragma unroll
    for (int m = 1; m < 16; m <<= 1)
#pragma unroll
      for (int r = 0; r < 8; ++r) ps[r] += __shfl_xor(ps[r], m, 32);
#pragma unroll
    for (int r = 0; r < 8; ++r) rl[r] = rl[r] * sc[r] + ps[r];
#pragma unroll
    for (int jd = 0; jd < 4; ++jd)
#pragma unroll
      for (int r = 0; r < 8; ++r) acc[jd][r] *= sc[r];

    // Re-layout P (C-fragment) -> LDS -> A-fragment (wave-local; LDS in-order).
    {
      _Float16* pb  = pbuf[wave];
      const int mhi = (lane >> 4) * 8;
      const int nl  = lane & 15;
#pragma unroll
      for (int r = 0; r < 8; ++r) {
        pb[(mhi + r) * 32 + nl]      = (_Float16)p0[r];
        pb[(mhi + r) * 32 + 16 + nl] = (_Float16)p1[r];
      }
      const v16h ap = frag_ld(pb, 32, lane);
#pragma unroll
      for (int jd = 0; jd < 4; ++jd)
        acc[jd] = wmma_f16(ap, frag_ld(&vbuf[cur][jd * 16 * 32], 32, lane), acc[jd]);
    }
    __syncthreads();
  }

  // Normalize and store ctx as [B,S,H*dh] f16.
  _Float16* obase = ctx + ((long)(b * SEQ + q0)) * MODEL_DIM + h * HEAD_DIM;
  const int mhi = (lane >> 4) * 8;
  const int nl  = lane & 15;
#pragma unroll
  for (int jd = 0; jd < 4; ++jd)
#pragma unroll
    for (int r = 0; r < 8; ++r)
      obase[(long)(mhi + r) * MODEL_DIM + jd * 16 + nl] =
          (_Float16)(acc[jd][r] / rl[r]);
}

// ---------------------------------------------------------------------------
// Host launcher
// ---------------------------------------------------------------------------
extern "C" void kernel_launch(void* const* d_in, const int* in_sizes, int n_in,
                              void* d_out, int out_size, void* d_ws, size_t ws_size,
                              hipStream_t stream) {
  (void)in_sizes; (void)n_in; (void)out_size; (void)ws_size;
  const float* q  = (const float*)d_in[0];
  const float* k  = (const float*)d_in[1];
  const float* v  = (const float*)d_in[2];
  const float* wq = (const float*)d_in[3];
  const float* wk = (const float*)d_in[4];
  const float* wv = (const float*)d_in[5];
  const float* wo = (const float*)d_in[6];

  // Workspace: qh, kh, V^T, ctx (4 x 16 MB) + 4 f16 weight copies (8 MB) = 72 MB.
  _Float16* qhb  = (_Float16*)d_ws;
  _Float16* khb  = qhb + (size_t)BS * MODEL_DIM;
  _Float16* vTb  = khb + (size_t)BS * MODEL_DIM;
  _Float16* ctx  = vTb + (size_t)BS * MODEL_DIM;
  _Float16* wq16 = ctx + (size_t)BS * MODEL_DIM;
  _Float16* wk16 = wq16 + (size_t)MODEL_DIM * MODEL_DIM;
  _Float16* wv16 = wk16 + (size_t)MODEL_DIM * MODEL_DIM;
  _Float16* wo16 = wv16 + (size_t)MODEL_DIM * MODEL_DIM;

  dim3 blk(256);
  const int wn = MODEL_DIM * MODEL_DIM;
  dim3 gcv(wn / (256 * 8));
  cvt_f32_f16<<<gcv, blk, 0, stream>>>(wq, wq16, wn);
  cvt_f32_f16<<<gcv, blk, 0, stream>>>(wk, wk16, wn);
  cvt_f32_f16<<<gcv, blk, 0, stream>>>(wv, wv16, wn);
  cvt_f32_f16<<<gcv, blk, 0, stream>>>(wo, wo16, wn);

  dim3 gp(MODEL_DIM / 128, BS / 128);
  gemm_wmma<float, false, true><<<gp, blk, 0, stream>>>(q, wq16, (void*)qhb, MODEL_DIM, MODEL_DIM);
  gemm_wmma<float, false, true><<<gp, blk, 0, stream>>>(k, wk16, (void*)khb, MODEL_DIM, MODEL_DIM);
  gemm_wmma<float, true,  true><<<gp, blk, 0, stream>>>(v, wv16, (void*)vTb, MODEL_DIM, MODEL_DIM);

  attn_kernel<<<dim3(SEQ / 128, BATCH * N_HEADS), blk, 0, stream>>>(qhb, khb, vTb, ctx);

  gemm_wmma<_Float16, false, false><<<gp, blk, 0, stream>>>(ctx, wo16, d_out, MODEL_DIM, MODEL_DIM);
}
